// MultiHeadGridAttention1d_3667902070981
// MI455X (gfx1250) — compile-verified, matrix operand order checked by reading, not run on hardware
//
#include <hip/hip_runtime.h>
#include <hip/hip_bf16.h>
#include <cstdint>
#include <cstddef>

// ---------------------------------------------------------------------------
// MultiHeadGridAttention1d for MI455X (gfx1250, wave32, WMMA + TDM)
// NH=8, KD=32, HD=64, C=512, b=4, w=20736 = 12^4  (W4 == w, no padding)
// ---------------------------------------------------------------------------

#define GC   512
#define GNH  8
#define GKD  32
#define GHD  64
#define GB   4
#define GW   20736      // 12^4
#define G12  12

typedef unsigned int u32;
typedef __attribute__((ext_vector_type(16))) __bf16 v16bf;
typedef __attribute__((ext_vector_type(8)))  float  v8f;
typedef __attribute__((ext_vector_type(4)))  u32    v4u;
typedef __attribute__((ext_vector_type(8)))  int    v8i;
typedef __attribute__((ext_vector_type(4)))  int    v4i;

#if defined(__has_builtin)
#if __has_builtin(__builtin_amdgcn_tensor_load_to_lds)
#define HAVE_TDM 1
#endif
#endif

#ifdef HAVE_TDM
// ---------------------------------------------------------------------------
// Tensor Data Mover: async 2D f32 tile load (tile_d1 rows x tile_d0 elems,
// row stride = row_stride elems) from global into LDS at byte offset lds_off.
// D# packing per CDNA5 ISA.
// ---------------------------------------------------------------------------
__device__ __forceinline__ void tdm_load_2d_f32(
    u32 lds_off, const void* gptr, u32 tile_d0, u32 tile_d1, uint64_t row_stride)
{
    const uint64_t ga = (uint64_t)(uintptr_t)gptr;
    v4u g0;
    g0[0] = 1u;                                           // count=1, user D#
    g0[1] = lds_off;                                      // lds_addr (bytes)
    g0[2] = (u32)(ga & 0xffffffffu);                      // global_addr[31:0]
    g0[3] = (u32)((ga >> 32) & 0x01ffffffu) | (2u << 30); // addr[56:32] | type=2
    v8i g1;
    g1[0] = (int)(2u << 16);                              // data_size=4B, no mask
    g1[1] = (int)((tile_d0 & 0xffffu) << 16);             // tensor_dim0[15:0]
    g1[2] = (int)((tile_d0 >> 16) | ((tile_d1 & 0xffffu) << 16)); // d0 hi | d1 lo
    g1[3] = (int)((tile_d1 >> 16) | (tile_d0 << 16));     // d1 hi | tile_dim0
    g1[4] = (int)(tile_d1 & 0xffffu);                     // tile_dim1 (dim2=0)
    g1[5] = (int)(u32)(row_stride & 0xffffffffu);         // dim0_stride[31:0]
    g1[6] = (int)(u32)((row_stride >> 32) & 0xffffu);     // dim0_stride[47:32]
    g1[7] = 0;
    v4i gz = {0, 0, 0, 0};
#if defined(__clang_major__) && (__clang_major__ >= 23)
    v8i gz8 = {0, 0, 0, 0, 0, 0, 0, 0};
    __builtin_amdgcn_tensor_load_to_lds(g0, g1, gz, gz, gz8, 0);
#else
    __builtin_amdgcn_tensor_load_to_lds(g0, g1, gz, gz, 0);
#endif
}
#endif // HAVE_TDM

// ---------------------------------------------------------------------------
// Fused conv1x1 GEMM:  out[b,m,n] = (sum_k W[m,k] * in[b,k,n]) * g[m] + beta[m]
// M = K = 512, N = GW per batch.
// Block: 256 thr (8 waves), tile BM=64 x BN=128.  Waves arranged 2(M) x 4(N),
// each wave owns a 32x32 sub-tile = 4 WMMA accumulators (2 A-frags x 2 B-frags).
// K stepped by 32 through double-buffered LDS bf16 tiles: ONE barrier / step.
// ---------------------------------------------------------------------------
#define BM 64
#define BN 128
#define TK 32
#define BSTRIDE (TK + 8)   // 40 bf16 = 80B row stride (16B aligned for b128 ds)

__global__ __launch_bounds__(256) void conv1x1_wmma_kernel(
    float* __restrict__ out, const float* __restrict__ in,
    const float* __restrict__ Wm, const float* __restrict__ gain,
    const float* __restrict__ beta, int Wd)
{
    __shared__ __bf16 As[2][BM][TK];        // weights (M x K)
    __shared__ __bf16 Bs[2][BN][BSTRIDE];   // activations transposed (N x K)

    const int b    = blockIdx.z;
    const int m0   = blockIdx.y * BM;
    const int n0   = blockIdx.x * BN;
    const int tid  = threadIdx.x;
    const int lane = tid & 31;
    const int wave = tid >> 5;
    const int wm0  = (wave >> 2) * 32;      // 2 wave-rows in M
    const int wn0  = (wave & 3) * 32;       // 4 wave-cols in N

    const float* inb = in + (size_t)b * GC * Wd;

    // staging decomposition (per thread)
    const int ar = tid >> 2;                // A row 0..63
    const int ac = (tid & 3) * 8;           // A col 0,8,16,24
    const int bk = tid >> 3;                // B k-row 0..31
    const int bn = (tid & 7) * 16;          // B n-col 0,16,...,112

    v8f acc[2][2];
    #pragma unroll
    for (int mi = 0; mi < 2; ++mi)
        #pragma unroll
        for (int ni = 0; ni < 2; ++ni)
            acc[mi][ni] = (v8f){0.f, 0.f, 0.f, 0.f, 0.f, 0.f, 0.f, 0.f};

    // prologue: stage k0 = 0 into buffer 0
    {
        const float* srcA = Wm + (size_t)(m0 + ar) * GC + ac;
        #pragma unroll
        for (int t = 0; t < 8; ++t) As[0][ar][ac + t] = (__bf16)srcA[t];
        const float* srcB = inb + (size_t)bk * Wd + n0 + bn;
        #pragma unroll
        for (int t = 0; t < 16; ++t) Bs[0][bn + t][bk] = (__bf16)srcB[t];
    }
    __syncthreads();

    const int half = lane >> 4;
    const int l15  = lane & 15;

    int buf = 0;
    for (int k0 = 0; k0 < GC; k0 += TK) {
        // stage next K tile into the other buffer while computing this one
        if (k0 + TK < GC) {
            const int nb = buf ^ 1;
            const float* srcA = Wm + (size_t)(m0 + ar) * GC + (k0 + TK) + ac;
            #pragma unroll
            for (int t = 0; t < 8; ++t) As[nb][ar][ac + t] = (__bf16)srcA[t];
            const float* srcB = inb + (size_t)(k0 + TK + bk) * Wd + n0 + bn;
            #pragma unroll
            for (int t = 0; t < 16; ++t) Bs[nb][bn + t][bk] = (__bf16)srcB[t];
            if (k0 + 2 * TK < GC) {
                __builtin_prefetch(Wm + (size_t)(m0 + ar) * GC + k0 + 2 * TK, 0, 1);
                __builtin_prefetch(inb + (size_t)(k0 + 2 * TK + bk) * Wd + n0, 0, 1);
            }
        }

        // fragments: lanes 0-15 hold K={0..7,16..23}, lanes 16-31 K={8..15,24..31}
        v16bf a0, a1, b0, b1;
        #pragma unroll
        for (int t = 0; t < 8; ++t) {
            a0[t]     = As[buf][wm0 + l15][half * 8 + t];
            a0[8 + t] = As[buf][wm0 + l15][16 + half * 8 + t];
            a1[t]     = As[buf][wm0 + 16 + l15][half * 8 + t];
            a1[8 + t] = As[buf][wm0 + 16 + l15][16 + half * 8 + t];
            b0[t]     = Bs[buf][wn0 + l15][half * 8 + t];
            b0[8 + t] = Bs[buf][wn0 + l15][16 + half * 8 + t];
            b1[t]     = Bs[buf][wn0 + 16 + l15][half * 8 + t];
            b1[8 + t] = Bs[buf][wn0 + 16 + l15][16 + half * 8 + t];
        }
        acc[0][0] = __builtin_amdgcn_wmma_f32_16x16x32_bf16(false, a0, false, b0, (short)0, acc[0][0], false, false);
        acc[0][1] = __builtin_amdgcn_wmma_f32_16x16x32_bf16(false, a0, false, b1, (short)0, acc[0][1], false, false);
        acc[1][0] = __builtin_amdgcn_wmma_f32_16x16x32_bf16(false, a1, false, b0, (short)0, acc[1][0], false, false);
        acc[1][1] = __builtin_amdgcn_wmma_f32_16x16x32_bf16(false, a1, false, b1, (short)0, acc[1][1], false, false);

        __syncthreads();
        buf ^= 1;
    }

    // Epilogue: D layout — VGPR r: lanes 0-15 -> M=r, lanes 16-31 -> M=8+r.
    #pragma unroll
    for (int mi = 0; mi < 2; ++mi) {
        #pragma unroll
        for (int ni = 0; ni < 2; ++ni) {
            const int n = n0 + wn0 + ni * 16 + l15;
            #pragma unroll
            for (int r = 0; r < 8; ++r) {
                const int m = m0 + wm0 + mi * 16 + r + half * 8;
                out[((size_t)b * GC + m) * Wd + n] = acc[mi][ni][r] * gain[m] + beta[m];
            }
        }
    }
}

// ---------------------------------------------------------------------------
// Head-sum of key channels: ksum[b,d,p] = sum_H qk[b, H*64 + 32 + d, p]
// ---------------------------------------------------------------------------
__global__ void ksum_kernel(float* __restrict__ out, const float* __restrict__ qk, int Wd)
{
    size_t i = (size_t)blockIdx.x * blockDim.x + threadIdx.x;
    const size_t tot = (size_t)GB * GKD * Wd;
    if (i >= tot) return;
    const int p = (int)(i % Wd);
    size_t r = i / Wd;
    const int d = (int)(r % GKD);
    const int b = (int)(r / GKD);
    float s = 0.f;
    #pragma unroll
    for (int H = 0; H < GNH; ++H)
        s += qk[((size_t)b * GC + H * 64 + 32 + d) * Wd + p];
    out[i] = s;
}

// ---------------------------------------------------------------------------
// Generic single-axis attention over a 12-long axis of the 12^4 grid.
//   score[P,p] = sum_d q[d, pos(p)] * ksum[d, pos(P)]   (softmax over p)
//   out[dv, pos(P)] = sum_p a[P,p] * v[dv, pos(p)]
// ---------------------------------------------------------------------------
__global__ __launch_bounds__(256) void attn_axis_kernel(
    float* __restrict__ outb, const float* __restrict__ qb,
    const float* __restrict__ ksum, const float* __restrict__ vb,
    int sa, int s0, int s1, int s2, int Wd)
{
    __shared__ float qs[GKD][G12];
    __shared__ float ks[GKD][G12];
    __shared__ float vs[GHD][G12];
    __shared__ float Sm[G12][G12 + 1];

    int sid = blockIdx.x;
    const int o2 = sid % G12; sid /= G12;
    const int o1 = sid % G12; sid /= G12;
    const int o0 = sid % G12; sid /= G12;
    const int H  = sid & 7;
    const int b  = sid >> 3;

    const size_t base = (size_t)o0 * s0 + (size_t)o1 * s1 + (size_t)o2 * s2;
    const size_t chq = ((size_t)b * GC + H * GHD) * Wd;
    const size_t chk = ((size_t)b * GKD) * Wd;
    const int tid = threadIdx.x;

    for (int t = tid; t < GKD * G12; t += 256) {
        const int d = t / G12, p = t % G12;
        qs[d][p] = qb[chq + (size_t)d * Wd + base + (size_t)p * sa];
        ks[d][p] = ksum[chk + (size_t)d * Wd + base + (size_t)p * sa];
    }
    for (int t = tid; t < GHD * G12; t += 256) {
        const int d = t / G12, p = t % G12;
        vs[d][p] = vb[chq + (size_t)d * Wd + base + (size_t)p * sa];
    }
    __syncthreads();

    for (int t = tid; t < G12 * G12; t += 256) {
        const int P = t / G12, p = t % G12;
        float s = 0.f;
        #pragma unroll
        for (int d = 0; d < GKD; ++d) s += qs[d][p] * ks[d][P];
        Sm[P][p] = s * 0.17677669529663687f;   // 1/sqrt(32)
    }
    __syncthreads();

    for (int P = tid; P < G12; P += 256) {
        float mx = -1e30f;
        #pragma unroll
        for (int p = 0; p < G12; ++p) mx = fmaxf(mx, Sm[P][p]);
        float sum = 0.f;
        #pragma unroll
        for (int p = 0; p < G12; ++p) { float e = __expf(Sm[P][p] - mx); Sm[P][p] = e; sum += e; }
        const float inv = 1.f / sum;
        #pragma unroll
        for (int p = 0; p < G12; ++p) Sm[P][p] *= inv;
    }
    __syncthreads();

    for (int t = tid; t < GHD * G12; t += 256) {
        const int d = t / G12, P = t % G12;
        float y = 0.f;
        #pragma unroll
        for (int p = 0; p < G12; ++p) y += Sm[P][p] * vs[d][p];
        outb[chq + (size_t)d * Wd + base + (size_t)P * sa] = y;
    }
}

// ---------------------------------------------------------------------------
// Final stage: per (b, H, I, J) block.
//   a3[K][k,l] = softmax_k( sum_d q2[d,k,l] * k2[d,K,l] * scale )
//   a4[K][L,l] = softmax_l( sum_d q2[d,K,l] * k2[d,K,L] * scale )
//   A34[L][k,l] = sum_K a3[K][k,l] * a4[K][L,l]
//   y[d, I,J,k,L] = sum_l t2[d, I,J,k,l] * A34[L][k,l]
// q2/k2 slices are 32x144 contiguous-row tiles -> loaded via the TDM.
// ---------------------------------------------------------------------------
__global__ __launch_bounds__(256) void attn_final_kernel(
    float* __restrict__ yb, const float* __restrict__ qb,
    const float* __restrict__ ksum, const float* __restrict__ t2b, int Wd)
{
    __shared__ float q2s[GKD][144];
    __shared__ float k2s[GKD][144];
    __shared__ float a3[G12][144];   // [K][k*12+l]
    __shared__ float a4[G12][144];   // [K][L*12+l]
    __shared__ float A34[G12][144];  // [L][k*12+l]

    int sid = blockIdx.x;
    const int J = sid % G12; sid /= G12;
    const int I = sid % G12; sid /= G12;
    const int H = sid & 7;
    const int b = sid >> 3;

    const size_t base = ((size_t)I * G12 + J) * 144;
    const size_t chq = ((size_t)b * GC + H * GHD) * Wd;
    const size_t chk = ((size_t)b * GKD) * Wd;
    const int tid = threadIdx.x;
    const float scale = 0.17677669529663687f;

#ifdef HAVE_TDM
    if ((tid >> 5) == 0) {   // wave 0 issues both DMA tiles (EXEC ignored by TDM)
        tdm_load_2d_f32((u32)(uintptr_t)&q2s[0][0], qb + chq + base, 144u, (u32)GKD, (uint64_t)Wd);
        tdm_load_2d_f32((u32)(uintptr_t)&k2s[0][0], ksum + chk + base, 144u, (u32)GKD, (uint64_t)Wd);
        __builtin_amdgcn_s_wait_tensorcnt(0);
    }
#else
    for (int t = tid; t < GKD * 144; t += 256) {
        const int d = t / 144, r = t % 144;
        q2s[d][r] = qb[chq + (size_t)d * Wd + base + r];
        k2s[d][r] = ksum[chk + (size_t)d * Wd + base + r];
    }
#endif
    __syncthreads();

    for (int t = tid; t < 1728; t += 256) {
        const int K = t / 144, r = t % 144;
        const int k = r / G12, l = r % G12;
        float s = 0.f;
        #pragma unroll
        for (int d = 0; d < GKD; ++d) s += q2s[d][k * G12 + l] * k2s[d][K * G12 + l];
        a3[K][r] = s * scale;
    }
    for (int t = tid; t < 1728; t += 256) {
        const int K = t / 144, r = t % 144;
        const int L = r / G12, l = r % G12;
        float s = 0.f;
        #pragma unroll
        for (int d = 0; d < GKD; ++d) s += q2s[d][K * G12 + l] * k2s[d][K * G12 + L];
        a4[K][r] = s * scale;
    }
    __syncthreads();

    // softmax a3 over k for fixed (K, l)
    for (int t = tid; t < 144; t += 256) {
        const int K = t / G12, l = t % G12;
        float mx = -1e30f;
        for (int k = 0; k < G12; ++k) mx = fmaxf(mx, a3[K][k * G12 + l]);
        float sum = 0.f;
        for (int k = 0; k < G12; ++k) { float e = __expf(a3[K][k * G12 + l] - mx); a3[K][k * G12 + l] = e; sum += e; }
        const float inv = 1.f / sum;
        for (int k = 0; k < G12; ++k) a3[K][k * G12 + l] *= inv;
    }
    // softmax a4 over l for fixed (K, L)
    for (int t = tid; t < 144; t += 256) {
        const int K = t / G12, L = t % G12;
        float mx = -1e30f;
        for (int l = 0; l < G12; ++l) mx = fmaxf(mx, a4[K][L * G12 + l]);
        float sum = 0.f;
        for (int l = 0; l < G12; ++l) { float e = __expf(a4[K][L * G12 + l] - mx); a4[K][L * G12 + l] = e; sum += e; }
        const float inv = 1.f / sum;
        for (int l = 0; l < G12; ++l) a4[K][L * G12 + l] *= inv;
    }
    __syncthreads();

    for (int t = tid; t < 1728; t += 256) {
        const int L = t / 144, r = t % 144;
        const int l = r % G12;
        float s = 0.f;
        #pragma unroll
        for (int K = 0; K < G12; ++K) s += a3[K][r] * a4[K][L * G12 + l];
        A34[L][r] = s;
    }
    __syncthreads();

    for (int t = tid; t < GHD * 144; t += 256) {
        const int d = t / 144, r = t % 144;
        const int k = r / G12, L = r % G12;
        float s = 0.f;
        #pragma unroll
        for (int l = 0; l < G12; ++l)
            s += t2b[chq + (size_t)d * Wd + base + k * G12 + l] * A34[L][k * G12 + l];
        yb[chq + (size_t)d * Wd + base + k * G12 + L] = s;
    }
}

// ---------------------------------------------------------------------------
// y += depthwise conv3(v0) * pe_g + pe_b   (cross-correlation, pad 1)
// ---------------------------------------------------------------------------
__global__ void pe_add_kernel(float* __restrict__ y, const float* __restrict__ v0,
                              const float* __restrict__ pw, const float* __restrict__ pg,
                              const float* __restrict__ pb, int Wd)
{
    size_t i = (size_t)blockIdx.x * blockDim.x + threadIdx.x;
    const size_t tot = (size_t)GB * GC * Wd;
    if (i >= tot) return;
    const int p = (int)(i % Wd);
    const int c = (int)((i / Wd) % GC);
    const float* v = v0 + (i - p);
    float s = v[p] * pw[c * 3 + 1];
    if (p > 0)      s += v[p - 1] * pw[c * 3 + 0];
    if (p < Wd - 1) s += v[p + 1] * pw[c * 3 + 2];
    y[i] += s * pg[c] + pb[c];
}

// ---------------------------------------------------------------------------
// Host launch
// ---------------------------------------------------------------------------
extern "C" void kernel_launch(void* const* d_in, const int* in_sizes, int n_in,
                              void* d_out, int out_size, void* d_ws, size_t ws_size,
                              hipStream_t stream)
{
    (void)in_sizes; (void)n_in; (void)out_size; (void)ws_size;

    const float* x      = (const float*)d_in[0];
    const float* qk1_w  = (const float*)d_in[1];
    const float* qk1_g  = (const float*)d_in[2];
    const float* qk1_b  = (const float*)d_in[3];
    const float* qk2_w  = (const float*)d_in[4];
    const float* qk2_g  = (const float*)d_in[5];
    const float* qk2_b  = (const float*)d_in[6];
    const float* v_w    = (const float*)d_in[7];
    const float* v_g    = (const float*)d_in[8];
    const float* v_b    = (const float*)d_in[9];
    const float* pe_w   = (const float*)d_in[10];
    const float* pe_g   = (const float*)d_in[11];
    const float* pe_b   = (const float*)d_in[12];
    const float* proj_w = (const float*)d_in[13];
    const float* proj_g = (const float*)d_in[14];
    const float* proj_b = (const float*)d_in[15];
    float* out = (float*)d_out;

    const int Wd = GW;
    const size_t big   = (size_t)GB * GC * Wd;     // 42,467,328 floats
    const size_t small = (size_t)GB * GKD * Wd;    //  2,654,208 floats

    float* ws0  = (float*)d_ws;                    // needs ~701 MB of workspace
    float* bufA = ws0;                             // qk1, later t2
    float* bufB = ws0 + big;                       // qk2
    float* bufC = ws0 + 2 * big;                   // v0
    float* bufD = ws0 + 3 * big;                   // t1, later y
    float* k1s  = ws0 + 4 * big;
    float* k2s  = k1s + small;

    const dim3 gGemm(Wd / BN, GC / BM, GB);        // (162, 8, 4)
    const dim3 bGemm(256);

    // 1-3: the three input conv1x1 GEMMs
    conv1x1_wmma_kernel<<<gGemm, bGemm, 0, stream>>>(bufA, x, qk1_w, qk1_g, qk1_b, Wd);
    conv1x1_wmma_kernel<<<gGemm, bGemm, 0, stream>>>(bufB, x, qk2_w, qk2_g, qk2_b, Wd);
    conv1x1_wmma_kernel<<<gGemm, bGemm, 0, stream>>>(bufC, x, v_w, v_g, v_b, Wd);

    // 4: head-summed keys
    {
        const int nb = (int)((small + 255) / 256);
        ksum_kernel<<<nb, 256, 0, stream>>>(k1s, bufA, Wd);
        ksum_kernel<<<nb, 256, 0, stream>>>(k2s, bufB, Wd);
    }

    // grid (i,j,k,l) strides: 1728, 144, 12, 1
    const int nSlices = GB * GNH * G12 * G12 * G12;   // 55296

    // 5: t1 = attend along i  (others j,k,l)
    attn_axis_kernel<<<nSlices, 256, 0, stream>>>(bufD, bufA, k1s, bufC,
                                                  1728, 144, 12, 1, Wd);
    // 6: t2 = attend along j  (others I,k,l) — qk1 (bufA) is dead, reuse it
    attn_axis_kernel<<<nSlices, 256, 0, stream>>>(bufA, bufB, k2s, bufD,
                                                  144, 1728, 12, 1, Wd);
    // 7: y = final a3/a4 contraction — t1 (bufD) dead, reuse it
    attn_final_kernel<<<GB * GNH * G12 * G12, 256, 0, stream>>>(bufD, bufB, k2s, bufA, Wd);

    // 8: y += pe(v0)
    {
        const int nb = (int)((big + 255) / 256);
        pe_add_kernel<<<nb, 256, 0, stream>>>(bufD, bufC, pe_w, pe_g, pe_b, Wd);
    }

    // 9: out = proj conv1x1
    conv1x1_wmma_kernel<<<gGemm, bGemm, 0, stream>>>(out, bufD, proj_w, proj_g, proj_b, Wd);
}
